// ConvLayer_16071767621927
// MI455X (gfx1250) — matrix operand order repaired and checked by reading, not compiled, not run on hardware
//
#include <hip/hip_runtime.h>
#include <math.h>

#define NATOMS 50000
#define MNBR   12
#define AF     128
#define NBRF   64
#define KDIM   320          // 2*AF + NBRF
#define CDIM   256          // 2*AF
#define ROWS   (NATOMS*MNBR) // 600000
#define EPSV   1e-5f

typedef __attribute__((ext_vector_type(16))) _Float16 v16h;
typedef __attribute__((ext_vector_type(8)))  float    v8f;

// ---- ws layout (bytes) ----
// 0       : Wt  f16 [256][320]  (transposed, col-major weights)   163840 B
// 163840  : stats f32[832] : g_sum[0,256) g_sq[256,512) cnt[512] g2s[576,704) g2q[704,832)
// 167168  : nbr_sumed f32 [50000][128]                            25.6 MB
#define WS_STATS_OFF 163840
#define WS_NSUM_OFF  167168

__device__ __forceinline__ float softplus_f(float x) {
    return x > 20.f ? x : log1pf(__expf(x));
}
__device__ __forceinline__ float sigmoid_f(float x) {
    return 1.f / (1.f + __expf(-x));
}

// Build one f16 A-fragment (16x32 layout) from 2 runs of 8 consecutive f32.
// p points at this lane's K-run base; run1 lives at p+8 floats.
__device__ __forceinline__ v16h load_afrag(const float* __restrict__ p) {
    const float4* q = (const float4*)p;
    float4 a0 = q[0], a1 = q[1];   // K base..base+7
    float4 b0 = q[2], b1 = q[3];   // K base+8..base+15
    v16h af;
    af[0]=(_Float16)a0.x; af[1]=(_Float16)a0.y; af[2]=(_Float16)a0.z; af[3]=(_Float16)a0.w;
    af[4]=(_Float16)a1.x; af[5]=(_Float16)a1.y; af[6]=(_Float16)a1.z; af[7]=(_Float16)a1.w;
    af[8]=(_Float16)b0.x; af[9]=(_Float16)b0.y; af[10]=(_Float16)b0.z; af[11]=(_Float16)b0.w;
    af[12]=(_Float16)b1.x; af[13]=(_Float16)b1.y; af[14]=(_Float16)b1.z; af[15]=(_Float16)b1.w;
    return af;
}

// One wave computes a 32-row x 128-col tile of (A @ W): two A-fragments share
// every B-fragment (halves B vmem traffic per WMMA vs 16-row tiling).
// A-frag layout (16-bit A 16x32): lane&15 = row; lanes<16 hold K [0..16),
// lanes>=16 hold K [16..32), same order as B (per K-chunk of 32).
// B-frag from Wt[c][k]: lane&15 = col; lanes<16 K[0..16), lanes>=16 K[16..32).
__device__ __forceinline__ void gemm32x128(
    const float* __restrict__ atom_fea,
    const float* __restrict__ nbr_fea,
    const int*   __restrict__ nbr_idx,
    const _Float16* __restrict__ Wt,
    int R0, int c0, int lane, v8f acc0[8], v8f acc1[8])
{
    int lr = lane & 15;
    int row0 = R0 + lr;
    int row1 = R0 + 16 + lr;
    const float* self0 = atom_fea + (size_t)(row0 / MNBR) * AF;
    const float* self1 = atom_fea + (size_t)(row1 / MNBR) * AF;
    const float* nbrp0 = atom_fea + (size_t)nbr_idx[row0] * AF;
    const float* nbrp1 = atom_fea + (size_t)nbr_idx[row1] * AF;
    const float* edge0 = nbr_fea + (size_t)row0 * NBRF;
    const float* edge1 = nbr_fea + (size_t)row1 * NBRF;
    int koff = (lane >> 4) ? 16 : 0;
    const _Float16* wl = Wt + (size_t)(c0 + lr) * KDIM + ((lane >> 4) ? 16 : 0);

    for (int kc = 0; kc < 10; ++kc) {
        const float *s0, *s1; int kl;
        if (kc < 4)      { s0 = self0; s1 = self1; kl = kc * 32; }
        else if (kc < 8) { s0 = nbrp0; s1 = nbrp1; kl = (kc - 4) * 32; }
        else             { s0 = edge0; s1 = edge1; kl = (kc - 8) * 32; }
        v16h af0 = load_afrag(s0 + kl + koff);
        v16h af1 = load_afrag(s1 + kl + koff);
#pragma unroll
        for (int t = 0; t < 8; ++t) {
            v16h bf = *(const v16h*)(wl + (size_t)t * 16 * KDIM + kc * 32);
            acc0[t] = __builtin_amdgcn_wmma_f32_16x16x32_f16(
                false, af0, false, bf, (short)0, acc0[t], false, false);
            acc1[t] = __builtin_amdgcn_wmma_f32_16x16x32_f16(
                false, af1, false, bf, (short)0, acc1[t], false, false);
        }
    }
}

// ---- small prep kernels ----
__global__ void k_prep_w(const float* __restrict__ W, _Float16* __restrict__ Wt) {
    int k = blockIdx.x;     // 0..319
    int c = threadIdx.x;    // 0..255
    Wt[(size_t)c * KDIM + k] = (_Float16)W[(size_t)k * CDIM + c];
}

__global__ void k_zero(float* __restrict__ stats) {
    int t = blockIdx.x * 256 + threadIdx.x;
    if (t < 832) stats[t] = 0.f;
}

__global__ void k_count(const float* __restrict__ mask, float* __restrict__ g_cnt) {
    __shared__ float red[256];
    int tid = threadIdx.x;
    float s = 0.f;
    for (int i = blockIdx.x * 256 + tid; i < ROWS; i += gridDim.x * 256) s += mask[i];
    red[tid] = s;
    __syncthreads();
    for (int o = 128; o > 0; o >>= 1) {
        if (tid < o) red[tid] += red[tid + o];
        __syncthreads();
    }
    if (tid == 0) unsafeAtomicAdd(g_cnt, red[0]);
}

// ---- pass 1: GEMM + masked per-channel sums (for batchnorm 1) ----
// 192 threads = 6 waves = 3 row-blocks (32 rows) x 2 col-halves -> 96 rows/WG.
__global__ __launch_bounds__(192) void k_pass1(
    const float* __restrict__ atom_fea, const float* __restrict__ nbr_fea,
    const int* __restrict__ nbr_idx, const float* __restrict__ mask,
    const float* __restrict__ b_fc, const _Float16* __restrict__ Wt,
    float* __restrict__ g_sum, float* __restrict__ g_sq)
{
    __shared__ float ls[CDIM], lq[CDIM];
    int tid = threadIdx.x;
    for (int c = tid; c < CDIM; c += 192) { ls[c] = 0.f; lq[c] = 0.f; }
    __syncthreads();

    int wid = tid >> 5, lane = tid & 31;
    int rb = wid >> 1, ch = wid & 1;
    int R0 = blockIdx.x * 96 + rb * 32;
    int c0 = ch * 128;
    v8f acc0[8] = {}, acc1[8] = {};
    gemm32x128(atom_fea, nbr_fea, nbr_idx, Wt, R0, c0, lane, acc0, acc1);

    int roff = (lane >> 4) ? 8 : 0;       // C layout: lanes>=16 hold rows M=v+8
    const float4* mp0 = (const float4*)(mask + R0 + roff);
    const float4* mp1 = (const float4*)(mask + R0 + 16 + roff);
    float4 ma = mp0[0], mb = mp0[1], mc = mp1[0], md = mp1[1];
    float mk0[8] = {ma.x, ma.y, ma.z, ma.w, mb.x, mb.y, mb.z, mb.w};
    float mk1[8] = {mc.x, mc.y, mc.z, mc.w, md.x, md.y, md.z, md.w};

#pragma unroll
    for (int t = 0; t < 8; ++t) {
        int ct = c0 + t * 16 + (lane & 15);
        float bias = b_fc[ct];
        float s = 0.f, q = 0.f;
#pragma unroll
        for (int v = 0; v < 8; ++v) {
            float x0 = acc0[t][v] + bias;
            float x1 = acc1[t][v] + bias;
            float mx0 = mk0[v] * x0;
            float mx1 = mk1[v] * x1;
            s += mx0 + mx1;
            q += mx0 * x0 + mx1 * x1;
        }
        atomicAdd(&ls[ct], s);
        atomicAdd(&lq[ct], q);
    }
    __syncthreads();
    for (int c = tid; c < CDIM; c += 192) {
        unsafeAtomicAdd(&g_sum[c], ls[c]);
        unsafeAtomicAdd(&g_sq[c],  lq[c]);
    }
}

// ---- pass 2: GEMM again + BN1 + sigmoid*softplus*mask + neighbor sum + BN2 partial sums
// 96 rows per block = exactly 8 atoms -> per-atom reduction never leaves the WGP.
__global__ __launch_bounds__(192) void k_pass2(
    const float* __restrict__ atom_fea, const float* __restrict__ nbr_fea,
    const int* __restrict__ nbr_idx, const float* __restrict__ mask,
    const float* __restrict__ b_fc, const float* __restrict__ gamma1,
    const float* __restrict__ beta1, const _Float16* __restrict__ Wt,
    const float* __restrict__ g_sum, const float* __restrict__ g_sq,
    const float* __restrict__ g_cnt, float* __restrict__ nbr_sumed,
    float* __restrict__ g2s, float* __restrict__ g2q)
{
    __shared__ float core_sp[3][32][128];   // softplus(core) staging (48 KB)
    __shared__ float atom_sum[8][128];      // 4 KB
    __shared__ float c2s[128], c2q[128];
    int tid = threadIdx.x;
    for (int idx = tid; idx < 1024; idx += 192) ((float*)atom_sum)[idx] = 0.f;
    if (tid < 128) { c2s[tid] = 0.f; c2q[tid] = 0.f; }
    __syncthreads();

    int wid = tid >> 5, lane = tid & 31;
    int rb = wid >> 1, ch = wid & 1;      // 3 row-blocks x {filter, core}
    int R0 = blockIdx.x * 96 + rb * 32;
    int c0 = ch * 128;
    v8f acc0[8] = {}, acc1[8] = {};
    gemm32x128(atom_fea, nbr_fea, nbr_idx, Wt, R0, c0, lane, acc0, acc1);

    float rcnt = 1.f / g_cnt[0];
    int roff = (lane >> 4) ? 8 : 0;

    // batchnorm 1 in-register (masked rows are the only ones consumed downstream)
#pragma unroll
    for (int t = 0; t < 8; ++t) {
        int ct = c0 + t * 16 + (lane & 15);
        float mean = g_sum[ct] * rcnt;
        float var  = g_sq[ct] * rcnt - mean * mean;
        float gm = gamma1[ct] * rsqrtf(var + EPSV);
        float bt = beta1[ct];
        float bias = b_fc[ct];
#pragma unroll
        for (int v = 0; v < 8; ++v) {
            float x0 = acc0[t][v] + bias;
            float x1 = acc1[t][v] + bias;
            acc0[t][v] = (x0 - mean) * gm + bt;
            acc1[t][v] = (x1 - mean) * gm + bt;
        }
    }

    if (ch == 1) {  // core half: stage softplus(core) in LDS
#pragma unroll
        for (int t = 0; t < 8; ++t) {
            int cl = t * 16 + (lane & 15);
#pragma unroll
            for (int v = 0; v < 8; ++v) {
                core_sp[rb][v + roff][cl]      = softplus_f(acc0[t][v]);
                core_sp[rb][16 + v + roff][cl] = softplus_f(acc1[t][v]);
            }
        }
    }
    __syncthreads();
    if (ch == 0) {  // filter half: gate, mask, reduce over neighbors into per-atom sums
        const float4* mp0 = (const float4*)(mask + R0 + roff);
        const float4* mp1 = (const float4*)(mask + R0 + 16 + roff);
        float4 ma = mp0[0], mb = mp0[1], mc = mp1[0], md = mp1[1];
        float mk0[8] = {ma.x, ma.y, ma.z, ma.w, mb.x, mb.y, mb.z, mb.w};
        float mk1[8] = {mc.x, mc.y, mc.z, mc.w, md.x, md.y, md.z, md.w};
#pragma unroll
        for (int t = 0; t < 8; ++t) {
            int cl = t * 16 + (lane & 15);
#pragma unroll
            for (int v = 0; v < 8; ++v) {
                int r0l = rb * 32 + v + roff;           // 0..95 within block
                int r1l = r0l + 16;
                float g0 = mk0[v] * sigmoid_f(acc0[t][v]) * core_sp[rb][v + roff][cl];
                float g1 = mk1[v] * sigmoid_f(acc1[t][v]) * core_sp[rb][16 + v + roff][cl];
                atomicAdd(&atom_sum[r0l / MNBR][cl], g0);
                atomicAdd(&atom_sum[r1l / MNBR][cl], g1);
            }
        }
    }
    __syncthreads();
    int atom0 = blockIdx.x * 8;
    for (int idx = tid; idx < 1024; idx += 192) {
        int a = idx >> 7, c = idx & 127;
        float val = atom_sum[a][c];
        nbr_sumed[(size_t)(atom0 + a) * AF + c] = val;
        atomicAdd(&c2s[c], val);
        atomicAdd(&c2q[c], val * val);
    }
    __syncthreads();
    if (tid < 128) {
        unsafeAtomicAdd(&g2s[tid], c2s[tid]);
        unsafeAtomicAdd(&g2q[tid], c2q[tid]);
    }
}

// ---- pass 3: batchnorm 2 + residual softplus ----
__global__ __launch_bounds__(256) void k_pass3(
    const float* __restrict__ atom_fea, const float* __restrict__ nbr_sumed,
    const float* __restrict__ g2s, const float* __restrict__ g2q,
    const float* __restrict__ gamma2, const float* __restrict__ beta2,
    float* __restrict__ out)
{
    int idx = blockIdx.x * 256 + threadIdx.x;
    if (idx >= NATOMS * AF) return;
    int c = idx & (AF - 1);
    const float rn = 1.f / (float)NATOMS;
    float m2 = g2s[c] * rn;
    float v2 = g2q[c] * rn - m2 * m2;
    float y = (nbr_sumed[idx] - m2) * rsqrtf(v2 + EPSV) * gamma2[c] + beta2[c];
    out[idx] = softplus_f(atom_fea[idx] + y);
}

extern "C" void kernel_launch(void* const* d_in, const int* in_sizes, int n_in,
                              void* d_out, int out_size, void* d_ws, size_t ws_size,
                              hipStream_t stream) {
    const float* atom = (const float*)d_in[0];
    const float* nbr  = (const float*)d_in[1];
    const int*   idx  = (const int*)d_in[2];
    const float* mask = (const float*)d_in[3];
    const float* W    = (const float*)d_in[4];
    const float* b    = (const float*)d_in[5];
    const float* g1   = (const float*)d_in[6];
    const float* be1  = (const float*)d_in[7];
    const float* g2   = (const float*)d_in[8];
    const float* be2  = (const float*)d_in[9];
    float* out = (float*)d_out;

    char* ws = (char*)d_ws;
    _Float16* Wt = (_Float16*)ws;
    float* stats = (float*)(ws + WS_STATS_OFF);
    float* g_sum = stats;
    float* g_sq  = stats + 256;
    float* g_cnt = stats + 512;
    float* g2s   = stats + 576;
    float* g2q   = stats + 704;
    float* nsum  = (float*)(ws + WS_NSUM_OFF);

    hipLaunchKernelGGL(k_prep_w, dim3(KDIM), dim3(CDIM), 0, stream, W, Wt);
    hipLaunchKernelGGL(k_zero,   dim3(4),    dim3(256),  0, stream, stats);
    hipLaunchKernelGGL(k_count,  dim3(512),  dim3(256),  0, stream, mask, g_cnt);
    hipLaunchKernelGGL(k_pass1,  dim3(ROWS / 96), dim3(192), 0, stream,
                       atom, nbr, idx, mask, b, Wt, g_sum, g_sq);
    hipLaunchKernelGGL(k_pass2,  dim3(ROWS / 96), dim3(192), 0, stream,
                       atom, nbr, idx, mask, b, g1, be1, Wt,
                       g_sum, g_sq, g_cnt, nsum, g2s, g2q);
    hipLaunchKernelGGL(k_pass3,  dim3((NATOMS * AF) / 256), dim3(256), 0, stream,
                       atom, nsum, g2s, g2q, g2, be2, out);
}
